// hmt_grn_22703197127351
// MI455X (gfx1250) — matrix operand order repaired and checked by reading, not compiled, not run on hardware
//
#include <hip/hip_runtime.h>
#include <hip/hip_bf16.h>
#include <math.h>

typedef __bf16 bf16;
typedef __attribute__((ext_vector_type(16))) __bf16 v16bf;
typedef __attribute__((ext_vector_type(8)))  __bf16 v8bf;
typedef __attribute__((ext_vector_type(8)))  float  v8f;

union AFrag { v16bf v; v8bf h[2]; };

__device__ __forceinline__ bf16 f2bf(float x) { return (bf16)x; }

#define BROWPAD 40   // 32 K-elems + 8 pad (80B row stride -> conflict-free LDS)

// ---------------------------------------------------------------------------
// f32 -> bf16 streaming convert (weights), 4 elems/thread
// ---------------------------------------------------------------------------
__global__ void hmt_cvt_bf16(const float* __restrict__ in, bf16* __restrict__ out, int n4) {
    int g = blockIdx.x * blockDim.x + threadIdx.x;
    if (g >= n4) return;
    float4 v = *(const float4*)(in + (size_t)g * 4);
    bf16* o = out + (size_t)g * 4;
    o[0] = f2bf(v.x); o[1] = f2bf(v.y); o[2] = f2bf(v.z); o[3] = f2bf(v.w);
}

// ---------------------------------------------------------------------------
// Gather embedding rows (f32 table -> bf16 rows), 4 cols per thread.
// ---------------------------------------------------------------------------
__global__ void hmt_gather_bf16(const float* __restrict__ table,
                                const int* __restrict__ idx,
                                bf16* __restrict__ out,
                                int rows, int ldo, int coloff) {
    int g = blockIdx.x * blockDim.x + threadIdx.x;
    if (g >= rows * 32) return;
    int row = g >> 5;
    int c4  = (g & 31) * 4;
    const float* src = table + (size_t)idx[row] * 128 + c4;
    bf16* dst = out + (size_t)row * ldo + coloff + c4;
    float4 v = *(const float4*)src;
    dst[0] = f2bf(v.x); dst[1] = f2bf(v.y); dst[2] = f2bf(v.z); dst[3] = f2bf(v.w);
}

// ---------------------------------------------------------------------------
// GEMM: C[M,N] = A[M,K](bf16 rm) @ B[N,K](bf16 rm)^T  via v_wmma bf16.
// Block = 256 threads = 8 waves stacked on M -> block tile 256M x 64N.
// B tile (64N x 32K) staged in LDS via async global->LDS DMA, double-buffered,
// shared by all 8 waves (8x L2-traffic reduction on B). A direct from global.
// ---------------------------------------------------------------------------
template<bool WF32, bool WBF16, bool ACCUM, bool BIAS>
__global__ void hmt_gemm_wmma(const bf16* __restrict__ A, int lda,
                              const bf16* __restrict__ B, int ldb,
                              const float* __restrict__ bias,
                              float* __restrict__ Cf,
                              bf16* __restrict__ Cbf, int ldc,
                              int M, int N, int K) {
    __shared__ __align__(16) bf16 Btile[2][64 * BROWPAD];

    const int lane = threadIdx.x & 31;
    const int wave = threadIdx.x >> 5;        // 0..7
    const int mr   = lane & 15;
    const int half = lane >> 4;
    const int n0   = blockIdx.x * 64;
    int m0 = (blockIdx.y * 8 + wave) * 32;
    if (m0 > M - 32) m0 = M - 32;             // M is always a multiple of 32

    int ar0 = m0 + mr;
    int ar1 = m0 + 16 + mr;
    const bf16* aptr0 = A + (size_t)ar0 * lda + 8 * half;
    const bf16* aptr1 = A + (size_t)ar1 * lda + 8 * half;

    // async fill of one 64x32 B tile: 256 threads x one b128 each
    const int fn  = (int)threadIdx.x >> 2;          // 0..63  (tile row / column n)
    const int fkq = ((int)threadIdx.x & 3) * 8;     // 0,8,16,24
    int fnn = n0 + fn; if (fnn >= N) fnn = N - 1;
    const bf16* fgp_base = B + (size_t)fnn * ldb + fkq;

    v8f acc[2][4];
    v8f zero = {0.f, 0.f, 0.f, 0.f, 0.f, 0.f, 0.f, 0.f};
#pragma unroll
    for (int i = 0; i < 2; ++i)
#pragma unroll
        for (int s = 0; s < 4; ++s) acc[i][s] = zero;

    const int nsteps = K / 32;

    // prologue: stage first tile
    {
        unsigned lo = (unsigned)(unsigned long long)&Btile[0][fn * BROWPAD + fkq];
        const bf16* gp = fgp_base;
        asm volatile("global_load_async_to_lds_b128 %0, %1, off"
                     :: "v"(lo), "v"(gp) : "memory");
    }

    for (int i = 0; i < nsteps; ++i) {
        __syncthreads();   // prior reads of buffer (i+1)&1 are complete
        if (i + 1 < nsteps) {
            unsigned lo = (unsigned)(unsigned long long)
                &Btile[(i + 1) & 1][fn * BROWPAD + fkq];
            const bf16* gp = fgp_base + (i + 1) * 32;
            asm volatile("global_load_async_to_lds_b128 %0, %1, off"
                         :: "v"(lo), "v"(gp) : "memory");
            asm volatile("s_wait_asynccnt 1" ::: "memory");   // fill(i) done
        } else {
            asm volatile("s_wait_asynccnt 0" ::: "memory");
        }
        __syncthreads();   // fill(i) visible to all waves

        const int k0 = i * 32;
        if (k0 + 32 < K) __builtin_prefetch(aptr0 + k0 + 32, 0, 3);

        AFrag a0, a1, b[4];
        a0.h[0] = *(const v8bf*)(aptr0 + k0);
        a0.h[1] = *(const v8bf*)(aptr0 + k0 + 16);
        a1.h[0] = *(const v8bf*)(aptr1 + k0);
        a1.h[1] = *(const v8bf*)(aptr1 + k0 + 16);
        const bf16* bbase = &Btile[i & 1][0];
#pragma unroll
        for (int s = 0; s < 4; ++s) {
            const bf16* bp = bbase + (s * 16 + mr) * BROWPAD + 8 * half;
            b[s].h[0] = *(const v8bf*)bp;
            b[s].h[1] = *(const v8bf*)(bp + 16);
        }
#pragma unroll
        for (int s = 0; s < 4; ++s) {
            acc[0][s] = __builtin_amdgcn_wmma_f32_16x16x32_bf16(
                false, a0.v, false, b[s].v, (short)0, acc[0][s], false, false);
            acc[1][s] = __builtin_amdgcn_wmma_f32_16x16x32_bf16(
                false, a1.v, false, b[s].v, (short)0, acc[1][s], false, false);
        }
    }

#pragma unroll
    for (int s = 0; s < 4; ++s) {
        int col = n0 + s * 16 + mr;
        if (col < N) {
            float bv = BIAS ? bias[col] : 0.f;
#pragma unroll
            for (int i = 0; i < 2; ++i) {
#pragma unroll
                for (int r = 0; r < 8; ++r) {
                    int row = m0 + 16 * i + r + 8 * half;
                    size_t o = (size_t)row * ldc + col;
                    float v = acc[i][s][r] + bv;
                    if (ACCUM) v += Cf[o];
                    if (WF32)  Cf[o]  = v;
                    if (WBF16) Cbf[o] = f2bf(v);
                }
            }
        }
    }
}

// ---------------------------------------------------------------------------
// GAT attention (bf16 in, bf16 out; math in f32):
// leaky(e_px + e_pn[n]) -> softmax over n -> weighted sum of pn.
// ---------------------------------------------------------------------------
__global__ void hmt_gat_attn(const bf16* __restrict__ e_px,
                             const bf16* __restrict__ e_pn,
                             const bf16* __restrict__ pn,
                             bf16* __restrict__ out, int BT) {
    int g = blockIdx.x * blockDim.x + threadIdx.x;
    if (g >= BT * 128) return;
    int bt = g >> 7;
    int f  = g & 127;
    float ex = (float)e_px[g];
    float vals[16];
    float mx = -1e30f;
#pragma unroll
    for (int n = 0; n < 16; ++n) {
        float e = ex + (float)e_pn[(size_t)(bt * 16 + n) * 128 + f];
        e = (e >= 0.f) ? e : 0.01f * e;        // LeakyReLU(0.01)
        vals[n] = e;
        mx = fmaxf(mx, e);
    }
    float ssum = 0.f;
#pragma unroll
    for (int n = 0; n < 16; ++n) { vals[n] = __expf(vals[n] - mx); ssum += vals[n]; }
    float inv = 1.f / ssum;
    float o = 0.f;
#pragma unroll
    for (int n = 0; n < 16; ++n)
        o += vals[n] * inv * (float)pn[(size_t)(bt * 16 + n) * 128 + f];
    out[g] = f2bf(o);
}

// ---------------------------------------------------------------------------
// LSTM: one WG per batch row. U (512x128 f32 = 256KB) staged once into
// dynamic LDS via CDNA5 async global->LDS DMA, then 64 recurrent steps read
// only LDS. h in static LDS; gates f32.
// ---------------------------------------------------------------------------
__global__ void hmt_lstm_seq(const float* __restrict__ G,
                             const float* __restrict__ U,
                             bf16* __restrict__ fused, int T) {
    extern __shared__ float Ush[];             // 65536 floats = 256 KB
    __shared__ float hsh[128];
    const int j = threadIdx.x;                 // 0..127

    for (int i = j * 4; i < 512 * 128; i += 128 * 4) {
        unsigned ldsoff = (unsigned)(unsigned long long)(&Ush[i]);
        const float* gp = U + i;
        asm volatile("global_load_async_to_lds_b128 %0, %1, off"
                     :: "v"(ldsoff), "v"(gp) : "memory");
    }
    asm volatile("s_wait_asynccnt 0" ::: "memory");
    hsh[j] = 0.f;
    float c = 0.f;
    __syncthreads();

    const int b = blockIdx.x;
    for (int t = 0; t < T; ++t) {
        size_t row = (size_t)b * T + t;
        const float* g = G + row * 512;
        float pre[4];
#pragma unroll
        for (int k = 0; k < 4; ++k) {
            const float* ur = &Ush[(k * 128 + j) * 128];
            float s = g[k * 128 + j];
            for (int e = 0; e < 128; ++e) s += ur[e] * hsh[e];
            pre[k] = s;
        }
        float ig = 1.f / (1.f + __expf(-pre[0]));
        float fg = 1.f / (1.f + __expf(-pre[1]));
        float og = 1.f / (1.f + __expf(-pre[2]));
        float gg = tanhf(pre[3]);
        c = fg * c + ig * gg;
        float h = og * tanhf(c);
        __syncthreads();
        hsh[j] = h;
        __syncthreads();
        fused[row * 256 + 128 + j] = f2bf(h);
    }
}

// ---------------------------------------------------------------------------
extern "C" void kernel_launch(void* const* d_in, const int* in_sizes, int n_in,
                              void* d_out, int out_size, void* d_ws, size_t ws_size,
                              hipStream_t stream) {
    const int B = 32, T = 64;
    const int BT = B * T;            // 2048
    const int BTN = BT * 16;         // 32768
    const int NCLS = 100000;

    const int*   x         = (const int*)d_in[0];
    const int*   users     = (const int*)d_in[1];
    const int*   sp_neigh  = (const int*)d_in[2];
    const int*   tp_neigh  = (const int*)d_in[3];
    const float* poi       = (const float*)d_in[4];
    const float* user_tab  = (const float*)d_in[5];
    const float* sp_w      = (const float*)d_in[6];
    const float* sp_aW     = (const float*)d_in[7];
    const float* sp_ab     = (const float*)d_in[8];
    const float* tp_w      = (const float*)d_in[9];
    const float* tp_aW     = (const float*)d_in[10];
    const float* tp_ab     = (const float*)d_in[11];
    const float* W         = (const float*)d_in[12];
    const float* U         = (const float*)d_in[13];
    const float* Ub        = (const float*)d_in[14];
    const float* sW        = (const float*)d_in[15];
    const float* tW        = (const float*)d_in[16];
    const float* fuseW     = (const float*)d_in[17];
    const float* fuseb     = (const float*)d_in[18];
    float* out = (float*)d_out;

    // workspace carve-up (256B aligned)
    char* wsp = (char*)d_ws;
    size_t off = 0;
    auto carve = [&](size_t bytes) -> void* {
        void* p = wsp + off;
        off = (off + bytes + 255) & ~(size_t)255;
        return p;
    };
    bf16*  Xbf     = (bf16*)carve((size_t)BT * 128 * 2);
    bf16*  NEIGHbf = (bf16*)carve((size_t)BTN * 128 * 2);
    bf16*  pxbf    = (bf16*)carve((size_t)BT * 128 * 2);
    bf16*  pnbf    = (bf16*)carve((size_t)BTN * 128 * 2);
    bf16*  epnbf   = (bf16*)carve((size_t)BTN * 128 * 2);
    bf16*  epxbf   = (bf16*)carve((size_t)BT * 128 * 2);
    bf16*  spatbf  = (bf16*)carve((size_t)BT * 128 * 2);
    bf16*  tempbf  = (bf16*)carve((size_t)BT * 128 * 2);
    float* G       = (float*)carve((size_t)BT * 512 * 4);
    bf16*  fusedbf = (bf16*)carve((size_t)BT * 256 * 2);
    // bf16 weight copies
    bf16* sp_wb   = (bf16*)carve(128 * 128 * 2);
    bf16* sp_aWb  = (bf16*)carve(128 * 256 * 2);
    bf16* tp_wb   = (bf16*)carve(128 * 128 * 2);
    bf16* tp_aWb  = (bf16*)carve(128 * 256 * 2);
    bf16* Wb      = (bf16*)carve(512 * 128 * 2);
    bf16* sWb     = (bf16*)carve(512 * 128 * 2);
    bf16* tWb     = (bf16*)carve(512 * 128 * 2);
    bf16* fuseWb  = (bf16*)carve((size_t)NCLS * 256 * 2);

    auto cvt = [&](const float* src, bf16* dst, int n) {
        hmt_cvt_bf16<<<(n / 4 + 255) / 256, 256, 0, stream>>>(src, dst, n / 4);
    };
    cvt(sp_w, sp_wb, 128 * 128);   cvt(sp_aW, sp_aWb, 128 * 256);
    cvt(tp_w, tp_wb, 128 * 128);   cvt(tp_aW, tp_aWb, 128 * 256);
    cvt(W, Wb, 512 * 128);         cvt(sW, sWb, 512 * 128);
    cvt(tW, tWb, 512 * 128);       cvt(fuseW, fuseWb, NCLS * 256);

    auto grid = [](int M, int N) { return dim3((N + 63) / 64, (M + 255) / 256); };
    const dim3 blk(256);

    // embeddings
    hmt_gather_bf16<<<(BT * 32 + 255) / 256, 256, 0, stream>>>(poi, x, Xbf, BT, 128, 0);
    hmt_gather_bf16<<<(BT * 32 + 255) / 256, 256, 0, stream>>>(user_tab, users, fusedbf, BT, 256, 0);

    // ---- spatial GAT ----
    hmt_gather_bf16<<<(BTN * 32 + 255) / 256, 256, 0, stream>>>(poi, sp_neigh, NEIGHbf, BTN, 128, 0);
    hmt_gemm_wmma<false, true, false, false><<<grid(BT, 128), blk, 0, stream>>>(
        Xbf, 128, sp_wb, 128, nullptr, nullptr, pxbf, 128, BT, 128, 128);
    hmt_gemm_wmma<false, true, false, false><<<grid(BTN, 128), blk, 0, stream>>>(
        NEIGHbf, 128, sp_wb, 128, nullptr, nullptr, pnbf, 128, BTN, 128, 128);
    hmt_gemm_wmma<false, true, false, false><<<grid(BT, 128), blk, 0, stream>>>(
        pxbf, 128, sp_aWb, 256, nullptr, nullptr, epxbf, 128, BT, 128, 128);
    hmt_gemm_wmma<false, true, false, true><<<grid(BTN, 128), blk, 0, stream>>>(
        pnbf, 128, sp_aWb + 128, 256, sp_ab, nullptr, epnbf, 128, BTN, 128, 128);
    hmt_gat_attn<<<(BT * 128 + 255) / 256, 256, 0, stream>>>(epxbf, epnbf, pnbf, spatbf, BT);

    // ---- temporal GAT (reuse buffers) ----
    hmt_gather_bf16<<<(BTN * 32 + 255) / 256, 256, 0, stream>>>(poi, tp_neigh, NEIGHbf, BTN, 128, 0);
    hmt_gemm_wmma<false, true, false, false><<<grid(BT, 128), blk, 0, stream>>>(
        Xbf, 128, tp_wb, 128, nullptr, nullptr, pxbf, 128, BT, 128, 128);
    hmt_gemm_wmma<false, true, false, false><<<grid(BTN, 128), blk, 0, stream>>>(
        NEIGHbf, 128, tp_wb, 128, nullptr, nullptr, pnbf, 128, BTN, 128, 128);
    hmt_gemm_wmma<false, true, false, false><<<grid(BT, 128), blk, 0, stream>>>(
        pxbf, 128, tp_aWb, 256, nullptr, nullptr, epxbf, 128, BT, 128, 128);
    hmt_gemm_wmma<false, true, false, true><<<grid(BTN, 128), blk, 0, stream>>>(
        pnbf, 128, tp_aWb + 128, 256, tp_ab, nullptr, epnbf, 128, BTN, 128, 128);
    hmt_gat_attn<<<(BT * 128 + 255) / 256, 256, 0, stream>>>(epxbf, epnbf, pnbf, tempbf, BT);

    // ---- LSTM pre-gate projections: G = X@W^T + Ub + spat@sW^T + temp@tW^T ----
    hmt_gemm_wmma<true, false, false, true><<<grid(BT, 512), blk, 0, stream>>>(
        Xbf, 128, Wb, 128, Ub, G, nullptr, 512, BT, 512, 128);
    hmt_gemm_wmma<true, false, true, false><<<grid(BT, 512), blk, 0, stream>>>(
        spatbf, 128, sWb, 128, nullptr, G, nullptr, 512, BT, 512, 128);
    hmt_gemm_wmma<true, false, true, false><<<grid(BT, 512), blk, 0, stream>>>(
        tempbf, 128, tWb, 128, nullptr, G, nullptr, 512, BT, 512, 128);

    // ---- sequential LSTM: U staged to LDS via async DMA (256 KB dynamic) ----
    hmt_lstm_seq<<<B, 128, 512 * 128 * sizeof(float), stream>>>(G, U, fusedbf, T);

    // ---- head: [2048,256] @ fuseWb[100000,256]^T + fuseb -> d_out ----
    hmt_gemm_wmma<true, false, false, true><<<grid(BT, NCLS), blk, 0, stream>>>(
        fusedbf, 256, fuseWb, 256, fuseb, out, nullptr, NCLS, BT, NCLS, 256);
}
